// DQNNet_84112639525158
// MI455X (gfx1250) — matrix operand order repaired and checked by reading, not compiled, not run on hardware
//
#include <hip/hip_runtime.h>
#include <hip/hip_bf16.h>

typedef __attribute__((ext_vector_type(16))) __bf16 v16bf;
typedef __attribute__((ext_vector_type(8)))  float  v8f;
typedef __attribute__((ext_vector_type(4)))  int    v4i_t;

#define D_FEAT 256
#define N_NODES 100000
#define N_EDGES 1600000

#if __has_builtin(__builtin_amdgcn_global_load_async_to_lds_b128) && \
    __has_builtin(__builtin_amdgcn_s_wait_asynccnt)
#define HAS_ASYNC_LDS 1
typedef __attribute__((address_space(1))) v4i_t* gptr_b128;   // global-AS int4*
typedef __attribute__((address_space(3))) v4i_t* lptr_b128;   // LDS-AS int4*
#else
#define HAS_ASYNC_LDS 0
#endif

__device__ __forceinline__ unsigned short f32_to_bf16_rte(float f) {
    unsigned int u = __float_as_uint(f);
    unsigned int r = u + 0x7FFFu + ((u >> 16) & 1u);   // round-to-nearest-even
    return (unsigned short)(r >> 16);
}

// ---------------- f32 -> bf16 conversion (vectorized) ----------------
__global__ __launch_bounds__(256)
void cvt_f32_bf16_kernel(const float* __restrict__ in, unsigned short* __restrict__ out, int n4) {
    int i = blockIdx.x * blockDim.x + threadIdx.x;
    if (i >= n4) return;
    float4 v = ((const float4*)in)[i];
    ushort4 o;
    o.x = f32_to_bf16_rte(v.x);
    o.y = f32_to_bf16_rte(v.y);
    o.z = f32_to_bf16_rte(v.z);
    o.w = f32_to_bf16_rte(v.w);
    ((ushort4*)out)[i] = o;
}

// ---------------- zero fill ----------------
__global__ __launch_bounds__(256)
void zero_f32_kernel(float* __restrict__ p, int n4) {
    int i = blockIdx.x * blockDim.x + threadIdx.x;
    if (i >= n4) return;
    ((float4*)p)[i] = make_float4(0.f, 0.f, 0.f, 0.f);
}

// ---------------- bf16 WMMA GEMM: out = relu(X @ W^T + b) ----------------
// X: [rows][256] bf16 row-major, W: [OUTF][256] bf16 row-major, out: f32 [rows][ldout]
// Block = 256 threads (8 waves), computes 32 rows x 64 output features.
union FragAB { v16bf v; uint4 q[2]; };

__global__ __launch_bounds__(256)
void gemm_bf16_wmma_kernel(const unsigned short* __restrict__ X,
                           const unsigned short* __restrict__ W,
                           const float* __restrict__ bias,
                           float* __restrict__ out,
                           int ldout, int col_off)
{
    __shared__ unsigned short lx[32][264];          // +8 pad: stride 528B -> conflict-free b128
    const int tid      = threadIdx.x;
    const int row_base = blockIdx.x * 32;
    const int cf_base  = blockIdx.y * 64;           // base output feature of this block

    // stage 32x256 bf16 X tile into LDS
    {
        const int r0  = tid >> 5;                   // 0..7
        const int c16 = tid & 31;                   // 16B chunk within a row
#if HAS_ASYNC_LDS
        // CDNA5 async global->LDS (GLOBAL_LOAD_ASYNC_TO_LDS_B128): bypasses VGPRs, ASYNCcnt
        #pragma unroll
        for (int p = 0; p < 4; ++p) {
            const int r = r0 + p * 8;
            unsigned short* gp = const_cast<unsigned short*>(
                X + (size_t)(row_base + r) * D_FEAT + c16 * 8);
            __builtin_amdgcn_global_load_async_to_lds_b128(
                (gptr_b128)(v4i_t*)gp,
                (lptr_b128)(v4i_t*)(&lx[r][c16 * 8]),
                0, 0);
        }
        __builtin_amdgcn_s_wait_asynccnt(0);
#else
        #pragma unroll
        for (int p = 0; p < 4; ++p) {
            const int r = r0 + p * 8;
            const uint4 v = *(const uint4*)(X + (size_t)(row_base + r) * D_FEAT + c16 * 8);
            *(uint4*)(&lx[r][c16 * 8]) = v;
        }
#endif
    }
    __syncthreads();

    const int lane     = tid & 31;
    const int wave     = tid >> 5;
    const int row_tile = wave & 1;                  // 0..1
    const int col_tile = wave >> 1;                 // 0..3
    const int m        = lane & 15;
    const int half     = lane >> 4;                 // lane group (ISA A/B layout halves)
    const int colF     = cf_base + col_tile * 16 + m;   // output feature this lane stores

    const unsigned short* wrow = W + (size_t)colF * D_FEAT;
    const unsigned short* arow = &lx[row_tile * 16 + m][0];

    v8f acc = {};
    #pragma unroll
    for (int k0 = 0; k0 < D_FEAT; k0 += 32) {
        FragAB a, b;
        // A 16x32 bf16 (ISA 7.12.2): VGPR0-3 = K k0+8*half+0..7, VGPR4-7 = K k0+16+8*half+0..7
        a.q[0] = *(const uint4*)(arow + k0 + 8 * half);
        a.q[1] = *(const uint4*)(arow + k0 + 16 + 8 * half);
        // B 32x16 bf16: same K order as A (K k0+8*half+0..7, then k0+16+8*half+0..7), N = lane&15
        b.q[0] = *(const uint4*)(wrow + k0 + 8 * half);
        b.q[1] = *(const uint4*)(wrow + k0 + 16 + 8 * half);
        acc = __builtin_amdgcn_wmma_f32_16x16x32_bf16(false, a.v, false, b.v,
                                                      (short)0, acc, false, false);
    }

    // C/D layout: VGPR i -> M = i + 8*half, N = lane&15
    const float bv   = bias[colF];
    const int   row0 = row_base + row_tile * 16 + 8 * half;
    const int   ocol = col_off + colF;
    #pragma unroll
    for (int i = 0; i < 8; ++i) {
        float v = acc[i] + bv;
        v = v > 0.f ? v : 0.f;                      // fused ReLU
        out[(size_t)(row0 + i) * ldout + ocol] = v;
    }
}

// ---------------- row L2 normalize; writes normalized f32 + bf16 shadow ----------------
__global__ __launch_bounds__(256)
void l2norm_kernel(float* __restrict__ h, unsigned short* __restrict__ hbf, int nrows) {
    const int row = blockIdx.x * 8 + (threadIdx.x >> 5);
    if (row >= nrows) return;
    const int lane = threadIdx.x & 31;
    float* p = h + (size_t)row * D_FEAT + lane * 8;
    float4 v0 = *(float4*)p;
    float4 v1 = *(float4*)(p + 4);
    float s = v0.x * v0.x + v0.y * v0.y + v0.z * v0.z + v0.w * v0.w
            + v1.x * v1.x + v1.y * v1.y + v1.z * v1.z + v1.w * v1.w;
    #pragma unroll
    for (int o = 16; o > 0; o >>= 1) s += __shfl_xor(s, o, 32);
    const float rn = rsqrtf(s);
    v0.x *= rn; v0.y *= rn; v0.z *= rn; v0.w *= rn;
    v1.x *= rn; v1.y *= rn; v1.z *= rn; v1.w *= rn;
    *(float4*)p       = v0;
    *(float4*)(p + 4) = v1;
    unsigned short* q = hbf + (size_t)row * D_FEAT + lane * 8;
    ushort4 o0, o1;
    o0.x = f32_to_bf16_rte(v0.x); o0.y = f32_to_bf16_rte(v0.y);
    o0.z = f32_to_bf16_rte(v0.z); o0.w = f32_to_bf16_rte(v0.w);
    o1.x = f32_to_bf16_rte(v1.x); o1.y = f32_to_bf16_rte(v1.y);
    o1.z = f32_to_bf16_rte(v1.z); o1.w = f32_to_bf16_rte(v1.w);
    *(ushort4*)q       = o0;
    *(ushort4*)(q + 4) = o1;
}

// ---------------- edge scatter-add: nm[dst] += h[src] (one wave per edge) ----------------
__global__ __launch_bounds__(256)
void scatter_add_kernel(const float* __restrict__ h, const int* __restrict__ ei,
                        float* __restrict__ nm) {
    const int e = blockIdx.x * 8 + (threadIdx.x >> 5);
    if (e >= N_EDGES) return;
    const int lane = threadIdx.x & 31;
    const int src = ei[e];
    const int dst = ei[N_EDGES + e];
    const float* sp = h  + (size_t)src * D_FEAT + lane * 8;
    float*       dp = nm + (size_t)dst * D_FEAT + lane * 8;
    float4 v0 = *(const float4*)sp;
    float4 v1 = *(const float4*)(sp + 4);
    __hip_atomic_fetch_add(dp + 0, v0.x, __ATOMIC_RELAXED, __HIP_MEMORY_SCOPE_AGENT);
    __hip_atomic_fetch_add(dp + 1, v0.y, __ATOMIC_RELAXED, __HIP_MEMORY_SCOPE_AGENT);
    __hip_atomic_fetch_add(dp + 2, v0.z, __ATOMIC_RELAXED, __HIP_MEMORY_SCOPE_AGENT);
    __hip_atomic_fetch_add(dp + 3, v0.w, __ATOMIC_RELAXED, __HIP_MEMORY_SCOPE_AGENT);
    __hip_atomic_fetch_add(dp + 4, v1.x, __ATOMIC_RELAXED, __HIP_MEMORY_SCOPE_AGENT);
    __hip_atomic_fetch_add(dp + 5, v1.y, __ATOMIC_RELAXED, __HIP_MEMORY_SCOPE_AGENT);
    __hip_atomic_fetch_add(dp + 6, v1.z, __ATOMIC_RELAXED, __HIP_MEMORY_SCOPE_AGENT);
    __hip_atomic_fetch_add(dp + 7, v1.w, __ATOMIC_RELAXED, __HIP_MEMORY_SCOPE_AGENT);
}

// ---------------- head: out[n] = dot(u[n,:], Wd2) + bd2 ----------------
__global__ __launch_bounds__(256)
void head_dot_kernel(const float* __restrict__ u, const float* __restrict__ wd2,
                     const float* __restrict__ bd2, float* __restrict__ out, int nrows) {
    const int row = blockIdx.x * 8 + (threadIdx.x >> 5);
    if (row >= nrows) return;
    const int lane = threadIdx.x & 31;
    const float* p = u   + (size_t)row * D_FEAT + lane * 8;
    const float* w = wd2 + lane * 8;
    float4 a0 = *(const float4*)p,      a1 = *(const float4*)(p + 4);
    float4 w0 = *(const float4*)w,      w1 = *(const float4*)(w + 4);
    float s = a0.x * w0.x + a0.y * w0.y + a0.z * w0.z + a0.w * w0.w
            + a1.x * w1.x + a1.y * w1.y + a1.z * w1.z + a1.w * w1.w;
    #pragma unroll
    for (int o = 16; o > 0; o >>= 1) s += __shfl_xor(s, o, 32);
    if (lane == 0) out[row] = s + bd2[0];
}

extern "C" void kernel_launch(void* const* d_in, const int* in_sizes, int n_in,
                              void* d_out, int out_size, void* d_ws, size_t ws_size,
                              hipStream_t stream) {
    const float* x   = (const float*)d_in[0];
    const int*   ei  = (const int*)d_in[1];
    const float* W1  = (const float*)d_in[2];  const float* b1  = (const float*)d_in[3];
    const float* W2  = (const float*)d_in[4];  const float* b2  = (const float*)d_in[5];
    const float* W3  = (const float*)d_in[6];  const float* b3  = (const float*)d_in[7];
    const float* Wd1 = (const float*)d_in[8];  const float* bd1 = (const float*)d_in[9];
    const float* Wd2 = (const float*)d_in[10]; const float* bd2 = (const float*)d_in[11];
    float* out = (float*)d_out;

    const size_t NF = (size_t)N_NODES * D_FEAT;
    char* ws = (char*)d_ws;
    size_t off = 0;
    float*          h     = (float*)(ws + off);          off += NF * 4;
    unsigned short* hbf   = (unsigned short*)(ws + off); off += NF * 2;
    float*          nm    = (float*)(ws + off);          off += NF * 4;   // reused as Wd1 output
    unsigned short* nmbf  = (unsigned short*)(ws + off); off += NF * 2;
    unsigned short* W1bf  = (unsigned short*)(ws + off); off += 256 * 256 * 2;
    unsigned short* W2bf  = (unsigned short*)(ws + off); off += 128 * 256 * 2;
    unsigned short* W3bf  = (unsigned short*)(ws + off); off += 128 * 256 * 2;
    unsigned short* Wd1bf = (unsigned short*)(ws + off); off += 256 * 256 * 2;
    (void)ws_size; (void)in_sizes; (void)n_in; (void)out_size;

    const int nf4       = (int)(NF / 4);
    const int cvt_blk   = (nf4 + 255) / 256;
    const int gemm_rows = N_NODES / 32;                 // 100000 / 32 = 3125, exact

    // 1) weights -> bf16
    cvt_f32_bf16_kernel<<<(256*256/4 + 255) / 256, 256, 0, stream>>>(W1,  W1bf,  256*256/4);
    cvt_f32_bf16_kernel<<<(128*256/4 + 255) / 256, 256, 0, stream>>>(W2,  W2bf,  128*256/4);
    cvt_f32_bf16_kernel<<<(128*256/4 + 255) / 256, 256, 0, stream>>>(W3,  W3bf,  128*256/4);
    cvt_f32_bf16_kernel<<<(256*256/4 + 255) / 256, 256, 0, stream>>>(Wd1, Wd1bf, 256*256/4);

    // 2) input layer: h = relu(x @ W1^T + b1), then l2norm -> (h f32, hbf bf16)
    cvt_f32_bf16_kernel<<<cvt_blk, 256, 0, stream>>>(x, hbf, nf4);
    gemm_bf16_wmma_kernel<<<dim3(gemm_rows, 4), 256, 0, stream>>>(hbf, W1bf, b1, h, 256, 0);
    l2norm_kernel<<<N_NODES / 8, 256, 0, stream>>>(h, hbf, N_NODES);

    // 3) two message-passing layers
    for (int it = 0; it < 2; ++it) {
        zero_f32_kernel<<<cvt_blk, 256, 0, stream>>>(nm, nf4);
        scatter_add_kernel<<<N_EDGES / 8, 256, 0, stream>>>(h, ei, nm);
        cvt_f32_bf16_kernel<<<cvt_blk, 256, 0, stream>>>(nm, nmbf, nf4);
        // concat([x@W2^T+b2, nm@W3^T+b3]) with fused relu, written into h (f32, ld=256)
        gemm_bf16_wmma_kernel<<<dim3(gemm_rows, 2), 256, 0, stream>>>(hbf,  W2bf, b2, h, 256, 0);
        gemm_bf16_wmma_kernel<<<dim3(gemm_rows, 2), 256, 0, stream>>>(nmbf, W3bf, b3, h, 256, 128);
        l2norm_kernel<<<N_NODES / 8, 256, 0, stream>>>(h, hbf, N_NODES);
    }

    // 4) head: u = relu(h @ Wd1^T + bd1) into nm buffer, then out = u @ Wd2^T + bd2
    gemm_bf16_wmma_kernel<<<dim3(gemm_rows, 4), 256, 0, stream>>>(hbf, Wd1bf, bd1, nm, 256, 0);
    head_dot_kernel<<<N_NODES / 8, 256, 0, stream>>>(nm, Wd2, bd2, out, N_NODES);
}
